// Head_12137577579251
// MI455X (gfx1250) — compile-verified, matrix operand order checked by reading, not run on hardware
//
#include <hip/hip_runtime.h>
#include <cstdint>

// Problem dims (compile-time)
#define B_ 32
#define T_ 1024
#define C_ 768
#define H_ 64

typedef __bf16 bf16;
typedef bf16  v16bf __attribute__((ext_vector_type(16)));
typedef bf16  v8bf  __attribute__((ext_vector_type(8)));
typedef float v8f   __attribute__((ext_vector_type(8)));

#define WMMA_BF16(a, b, c) \
  __builtin_amdgcn_wmma_f32_16x16x32_bf16(false, (a), false, (b), (short)0, (c), false, false)

__device__ __forceinline__ v8f zero8() {
  v8f z = {0.f, 0.f, 0.f, 0.f, 0.f, 0.f, 0.f, 0.f};
  return z;
}

// ---------------------------------------------------------------------------
// Fragment loaders (wave32 WMMA 16-bit operand layout, ISA 7.12.2):
//   lane L: row/col index = L&15 ; k-base = (L&16) ? 8 : 0
//   vec elems [0..7]  -> K = kbase + 0..7 ; elems [8..15] -> K = kbase + 16..23
// Identical layout serves as A (row = M) or B (col = N), so one row-major
// 16x32 tile loader covers both operand roles.
// ---------------------------------------------------------------------------
__device__ __forceinline__ v16bf load_frag_f32(const float* __restrict__ base,
                                               int ld, int lane) {
  const float* p = base + (size_t)(lane & 15) * ld + ((lane & 16) ? 8 : 0);
  v8f a = *(const v8f*)p;         // 32B aligned at all call sites
  v8f b = *(const v8f*)(p + 16);
  v16bf f;
#pragma unroll
  for (int j = 0; j < 8; ++j) { f[j] = (bf16)a[j]; f[8 + j] = (bf16)b[j]; }
  return f;
}

__device__ __forceinline__ v16bf load_frag_bf16(const bf16* base, int ld, int lane) {
  const bf16* p = base + (size_t)(lane & 15) * ld + ((lane & 16) ? 8 : 0);
  v8bf a = *(const v8bf*)p;       // 16B aligned at all call sites
  v8bf b = *(const v8bf*)(p + 16);
  v16bf f;
#pragma unroll
  for (int j = 0; j < 8; ++j) { f[j] = a[j]; f[8 + j] = b[j]; }
  return f;
}

// Async global -> LDS copy of one 16B chunk (ASYNCcnt-tracked, ISA 10.x/§15.18)
__device__ __forceinline__ void async_ld16(uint32_t lds_byte_off, const bf16* gp) {
  asm volatile("global_load_async_to_lds_b128 %0, %1, off"
               :: "v"(lds_byte_off), "v"(gp) : "memory");
}

// ---------------------------------------------------------------------------
// Kernel 0: one-shot weight conversion f32 -> bf16 (Wq|Wk|Wv packed).
// Removes ~2300 v_cvt per projection wave and halves weight L2 traffic.
// ---------------------------------------------------------------------------
__global__ __launch_bounds__(256) void wcvt_kernel(
    const float* __restrict__ Wq, const float* __restrict__ Wk,
    const float* __restrict__ Wv, bf16* __restrict__ wb) {
  const int NW = H_ * C_;
  const int i = (blockIdx.x * 256 + threadIdx.x) * 8;
  if (i >= NW) return;
  const float* src[3] = {Wq, Wk, Wv};
#pragma unroll
  for (int w = 0; w < 3; ++w) {
    v8f a = *(const v8f*)(src[w] + i);
    v8bf o;
#pragma unroll
    for (int j = 0; j < 8; ++j) o[j] = (bf16)a[j];
    *(v8bf*)(wb + (size_t)w * NW + i) = o;
  }
}

// ---------------------------------------------------------------------------
// Kernel 1: QKV projection.  One wave per 16-row t-tile.
//   q[b,t,h] (bf16, pre-scaled by H^-0.5), k[b,t,h] (bf16), vT[b,h,t] (bf16)
// q/k tiles:  D[t,h] = A(x 16tx32c) * B(W^T 32cx16h)
// v  tiles:   D[h,t] = A(Wv 16hx32c) * B(x^T 32cx16t)   (same fragments!)
// ---------------------------------------------------------------------------
__global__ __launch_bounds__(128) void qkv_proj_kernel(
    const float* __restrict__ x, const bf16* __restrict__ wb,
    bf16* __restrict__ qo, bf16* __restrict__ ko, bf16* __restrict__ vTo) {
  const int lane  = threadIdx.x & 31;
  const int wid   = blockIdx.x * (blockDim.x >> 5) + (threadIdx.x >> 5);
  const int b     = wid >> 6;          // 64 t-tiles per batch
  const int tt    = wid & 63;
  const int tbase = tt * 16;

  const int NW = H_ * C_;
  const bf16* wq = wb;
  const bf16* wk = wb + NW;
  const bf16* wv = wb + 2 * NW;

  const float* xb = x + ((size_t)b * T_ + tbase) * C_;

  v8f accq[4], acck[4], accv[4];
#pragma unroll
  for (int i = 0; i < 4; ++i) { accq[i] = zero8(); acck[i] = zero8(); accv[i] = zero8(); }

  for (int c0 = 0; c0 < C_; c0 += 32) {
    if (c0 + 32 < C_) __builtin_prefetch(xb + c0 + 32, 0, 0);
    v16bf xf = load_frag_f32(xb + c0, C_, lane);
#pragma unroll
    for (int ht = 0; ht < 4; ++ht) {
      v16bf wqf = load_frag_bf16(wq + (size_t)(ht * 16) * C_ + c0, C_, lane);
      accq[ht] = WMMA_BF16(xf, wqf, accq[ht]);
      v16bf wkf = load_frag_bf16(wk + (size_t)(ht * 16) * C_ + c0, C_, lane);
      acck[ht] = WMMA_BF16(xf, wkf, acck[ht]);
      v16bf wvf = load_frag_bf16(wv + (size_t)(ht * 16) * C_ + c0, C_, lane);
      accv[ht] = WMMA_BF16(wvf, xf, accv[ht]);   // transposed: D[h,t]
    }
  }

  const int n  = lane & 15;
  const int mb = (lane & 16) ? 8 : 0;
#pragma unroll
  for (int ht = 0; ht < 4; ++ht) {
#pragma unroll
    for (int r = 0; r < 8; ++r) {
      const int trow = tbase + mb + r;
      qo[((size_t)b * T_ + trow) * H_ + ht * 16 + n] = (bf16)(accq[ht][r] * 0.125f);
      ko[((size_t)b * T_ + trow) * H_ + ht * 16 + n] = (bf16)(acck[ht][r]);
      vTo[((size_t)b * H_ + ht * 16 + mb + r) * T_ + tbase + n] = (bf16)(accv[ht][r]);
    }
  }
}

// ---------------------------------------------------------------------------
// Kernel 2: fused causal flash attention.  4 waves/block, same batch,
// adjacent 16-query tiles; K/V 32-key blocks are shared through LDS via
// double-buffered global_load_async_to_lds_b128 (ASYNCcnt).
// S^T = K*Q^T so each lane owns one query column; the two 16x16 S^T f32
// accumulator subtiles map element-for-element onto the 32x16 bf16 B-fragment
// of P^T (zero cross-lane traffic).  O^T = V^T * P^T accumulated in f32.
// ---------------------------------------------------------------------------
__global__ __launch_bounds__(128) void attn_kernel(
    const bf16* __restrict__ q, const bf16* __restrict__ k,
    const bf16* __restrict__ vT, float* __restrict__ y) {
  __shared__ bf16 sK[2][32 * 64];   // [buf][key][h]   4KB each
  __shared__ bf16 sV[2][64 * 32];   // [buf][h][key]   4KB each

  const int tid   = threadIdx.x;
  const int lane  = tid & 31;
  const int w     = tid >> 5;                  // wave index in block
  const int b     = blockIdx.x >> 4;           // 16 blocks per batch
  const int qt0   = (blockIdx.x & 15) * 4;     // first q-tile of this block
  const int qt    = qt0 + w;
  const int qbase = qt * 16;
  const int n     = lane & 15;
  const int mb    = (lane & 16) ? 8 : 0;
  const int gq    = qbase + n;                 // this lane's global query row

  const bf16* qb   = q  + (size_t)b * T_ * H_;
  const bf16* kptr = k  + (size_t)b * T_ * H_;
  const bf16* vptr = vT + (size_t)b * H_ * T_;

  const uint32_t offK[2] = {(uint32_t)(uintptr_t)&sK[0][0], (uint32_t)(uintptr_t)&sK[1][0]};
  const uint32_t offV[2] = {(uint32_t)(uintptr_t)&sV[0][0], (uint32_t)(uintptr_t)&sV[1][0]};

  // Q fragments (B role), h-chunks [0,32) and [32,64); softmax scale pre-baked.
  const v16bf qf0 = load_frag_bf16(qb + (size_t)qbase * H_ + 0, H_, lane);
  const v16bf qf1 = load_frag_bf16(qb + (size_t)qbase * H_ + 32, H_, lane);

  v8f o[4];
#pragma unroll
  for (int i = 0; i < 4; ++i) o[i] = zero8();
  float m_i = -1e30f, l_i = 0.f;

  const int my_nb  = (qt >> 1) + 1;            // 32-key blocks up to my diagonal
  const int max_nb = ((qt0 + 3) >> 1) + 1;     // block-uniform loop bound

  // Issue async fill of buffer 0 (K: 256 16B chunks, V: 256 16B chunks; 4/thread)
  {
#pragma unroll
    for (int j = 0; j < 2; ++j) {
      const int c = tid + j * 128;
      async_ld16(offK[0] + c * 16, kptr + (size_t)(c >> 3) * H_ + (c & 7) * 8);
      async_ld16(offV[0] + c * 16, vptr + (size_t)(c >> 2) * T_ + (c & 3) * 8);
    }
  }

  for (int kb_i = 0; kb_i < max_nb; ++kb_i) {
    const int buf   = kb_i & 1;
    const int kbase = kb_i * 32;

    if (kb_i + 1 < max_nb) {                   // prefetch next block into other buffer
      const int nb = kbase + 32;
#pragma unroll
      for (int j = 0; j < 2; ++j) {
        const int c = tid + j * 128;
        async_ld16(offK[buf ^ 1] + c * 16, kptr + (size_t)(nb + (c >> 3)) * H_ + (c & 7) * 8);
        async_ld16(offV[buf ^ 1] + c * 16, vptr + (size_t)(c >> 2) * T_ + nb + (c & 3) * 8);
      }
      asm volatile("s_wait_asynccnt 0x4" ::: "memory");  // current buffer done, next in flight
    } else {
      asm volatile("s_wait_asynccnt 0x0" ::: "memory");
    }
    __syncthreads();                           // all waves' chunks visible

    if (kb_i < my_nb) {                        // wave-uniform: EXEC stays all-ones
      const bf16* sKb = &sK[buf][0];
      const bf16* sVb = &sV[buf][0];

      // S^T subtiles from LDS: s0 = keys [0,16), s1 = keys [16,32) of block
      v8f s0 = zero8(), s1 = zero8();
      {
        v16bf kf;
        kf = load_frag_bf16(sKb + 0,            64, lane); s0 = WMMA_BF16(kf, qf0, s0);
        kf = load_frag_bf16(sKb + 32,           64, lane); s0 = WMMA_BF16(kf, qf1, s0);
        kf = load_frag_bf16(sKb + 16 * 64,      64, lane); s1 = WMMA_BF16(kf, qf0, s1);
        kf = load_frag_bf16(sKb + 16 * 64 + 32, 64, lane); s1 = WMMA_BF16(kf, qf1, s1);
      }

      if (kbase + 31 > qbase) {                // causal mask (diagonal block only)
#pragma unroll
        for (int r = 0; r < 8; ++r) {
          if (kbase + mb + r > gq)      s0[r] = -1e30f;
          if (kbase + 16 + mb + r > gq) s1[r] = -1e30f;
        }
      }

      // Online softmax (per query column; pair lane with lane^16)
      float mx = s0[0];
#pragma unroll
      for (int r = 0; r < 8; ++r) { mx = fmaxf(mx, s0[r]); mx = fmaxf(mx, s1[r]); }
      mx = fmaxf(mx, __shfl_xor(mx, 16, 32));
      const float m_new = fmaxf(m_i, mx);
      const float corr  = __expf(m_i - m_new);
      float rs = 0.f;
#pragma unroll
      for (int r = 0; r < 8; ++r) {
        s0[r] = __expf(s0[r] - m_new);
        s1[r] = __expf(s1[r] - m_new);
        rs += s0[r] + s1[r];
      }
      rs += __shfl_xor(rs, 16, 32);
      l_i = l_i * corr + rs;
      m_i = m_new;
#pragma unroll
      for (int ht = 0; ht < 4; ++ht)
#pragma unroll
        for (int r = 0; r < 8; ++r) o[ht][r] *= corr;

      // P^T B-fragment (32 keys x 16 queries): direct per-lane repack
      v16bf pf;
#pragma unroll
      for (int r = 0; r < 8; ++r) { pf[r] = (bf16)s0[r]; pf[8 + r] = (bf16)s1[r]; }

      // O^T[h,q] += V^T[h,keys] * P^T[keys,q]
#pragma unroll
      for (int ht = 0; ht < 4; ++ht) {
        v16bf vf = load_frag_bf16(sVb + ht * 16 * 32, 32, lane);
        o[ht] = WMMA_BF16(vf, pf, o[ht]);
      }
    }
    __syncthreads();                           // readers done before buffer reuse
  }

  const float inv = 1.0f / l_i;
  float* yb = y + ((size_t)b * T_ + gq) * H_;  // this lane's query row
#pragma unroll
  for (int ht = 0; ht < 4; ++ht) {
    float4 lo = make_float4(o[ht][0] * inv, o[ht][1] * inv, o[ht][2] * inv, o[ht][3] * inv);
    float4 hi = make_float4(o[ht][4] * inv, o[ht][5] * inv, o[ht][6] * inv, o[ht][7] * inv);
    *(float4*)(yb + ht * 16 + mb + 0) = lo;    // 32B-aligned (mb in {0,8})
    *(float4*)(yb + ht * 16 + mb + 4) = hi;
  }
}

// ---------------------------------------------------------------------------
extern "C" void kernel_launch(void* const* d_in, const int* in_sizes, int n_in,
                              void* d_out, int out_size, void* d_ws, size_t ws_size,
                              hipStream_t stream) {
  const float* x  = (const float*)d_in[0];
  const float* Wq = (const float*)d_in[1];
  const float* Wk = (const float*)d_in[2];
  const float* Wv = (const float*)d_in[3];
  float* y = (float*)d_out;

  // Workspace: q, k (bf16 [B,T,H]), vT (bf16 [B,H,T]), bf16 weights (3x[H,C])
  const size_t NQK = (size_t)B_ * T_ * H_;
  bf16* qo  = (bf16*)d_ws;
  bf16* ko  = qo + NQK;
  bf16* vTo = ko + NQK;
  bf16* wb  = vTo + NQK;                       // 3 * 48K bf16 = 288 KB

  wcvt_kernel<<<dim3((H_ * C_ / 8 + 255) / 256), dim3(256), 0, stream>>>(Wq, Wk, Wv, wb);

  const dim3 block(128);                       // 4 waves / block
  const dim3 grid(B_ * (T_ / 16) / 4);         // 2048 waves total
  qkv_proj_kernel<<<grid, block, 0, stream>>>(x, wb, qo, ko, vTo);
  attn_kernel<<<grid, block, 0, stream>>>(qo, ko, vTo, y);
}
